// MultiheadAttention_31619549233328
// MI455X (gfx1250) — compile-verified, matrix operand order checked by reading, not run on hardware
//
#include <hip/hip_runtime.h>

typedef __attribute__((ext_vector_type(16))) __bf16 v16bf;
typedef __attribute__((ext_vector_type(8)))  float  v8f;
typedef __attribute__((ext_vector_type(4)))  unsigned u32x4;
typedef __attribute__((ext_vector_type(8)))  int      i32x8;
typedef __attribute__((ext_vector_type(4)))  int      i32x4;

union Frag { unsigned u[8]; v16bf v; };

__device__ __forceinline__ unsigned short f2bf(float f) {
    unsigned u = __builtin_bit_cast(unsigned, f);
    u += 0x7FFFu + ((u >> 16) & 1u);          // round-to-nearest-even
    return (unsigned short)(u >> 16);
}
__device__ __forceinline__ unsigned pk(unsigned short lo, unsigned short hi) {
    return (unsigned)lo | ((unsigned)hi << 16);
}
// bf16 A/B operand K-offset pattern (16x32 tile, wave32):
// lanes 0-15: vgpr i holds K pair at {0,2,4,6,16,18,20,22}[i]; lanes 16-31: +8
__device__ __forceinline__ int koffs(int i, int lane) {
    return ((i < 4) ? (2 * i) : (2 * i + 8)) + ((lane >> 4) << 3);
}
__device__ __forceinline__ v8f wmma_bf16(const Frag& a, const Frag& b, v8f c) {
    return __builtin_amdgcn_wmma_f32_16x16x32_bf16(false, a.v, false, b.v,
                                                   (short)0, c, false, false);
}

// ---------------------------------------------------------------------------
// Tensor Data Mover: 2D bf16 tile load Global -> LDS (6-arg builtin form).
// D# group0: [1:0]=count=1, [63:32]=lds_addr, [120:64]=global byte addr,
//            [127:126]=type=2. group1: data_size=2B, tensor dims == tile dims,
//            tile_dim0 (contig) x tile_dim1 rows, dim0 stride in elements.
// ---------------------------------------------------------------------------
__device__ __forceinline__ void tdm_load_2d_bf16(
    unsigned ldsOff, const unsigned short* gptr,
    unsigned tileD0, unsigned tileD1, unsigned strideD0)
{
    unsigned long long ga = (unsigned long long)(uintptr_t)gptr;
    u32x4 g0;
    g0[0] = 1u;                                              // count = 1 (user D#)
    g0[1] = ldsOff;                                          // LDS byte address
    g0[2] = (unsigned)ga;                                    // global addr [95:64]
    g0[3] = (unsigned)((ga >> 32) & 0x01FFFFFFu) | (2u << 30); // [120:96] | type=2
    i32x8 g1;
    g1[0] = 0x00010000;                                      // data_size = 2 bytes
    g1[1] = (int)(tileD0 << 16);                             // tensor_dim0 lo16
    g1[2] = (int)(tileD1 << 16);                             // dim0 hi=0 | tensor_dim1 lo16
    g1[3] = (int)(tileD0 << 16);                             // dim1 hi=0 | tile_dim0
    g1[4] = (int)tileD1;                                     // tile_dim1 | tile_dim2=0
    g1[5] = (int)strideD0;                                   // tensor_dim0_stride lo32
    g1[6] = 0;
    g1[7] = 0;
    i32x4 zero4 = {0, 0, 0, 0};
    i32x8 zero8 = {0, 0, 0, 0, 0, 0, 0, 0};
    __builtin_amdgcn_tensor_load_to_lds(g0, g1, zero4, zero4, zero8, 0);
}

// ---------------------------------------------------------------------------
// GEMM: C[row,col] = sum_k A[row,k] * W[col,k] + bias[col]
// MODE 0: A = fp32 x; epilogue scatters bf16 q/(B,H,N,D), k/(B,H,N,D),
//         vT/(B,H,D,N); q pre-scaled by 1/sqrt(64).
// MODE 1: A = bf16 attention output; epilogue writes fp32 d_out.
// Block: 256 threads (8 waves), tile 128x128, BK=32.
// ---------------------------------------------------------------------------
template<int MODE>
__global__ __launch_bounds__(256) void gemm_wmma(
    const float* __restrict__ Af, const unsigned short* __restrict__ Ah,
    const float* __restrict__ W, const float* __restrict__ bias,
    unsigned short* __restrict__ qo, unsigned short* __restrict__ ko,
    unsigned short* __restrict__ vto, float* __restrict__ out)
{
    __shared__ unsigned sA[128 * 16];   // 128 rows x 32 bf16 (packed pairs)
    __shared__ unsigned sB[128 * 16];

    const int t = threadIdx.x;
    const int lane = t & 31, wave = t >> 5;
    const int wm = wave >> 1, wn = wave & 1;      // 4x2 wave grid
    const int mBase = blockIdx.y * 128, nBase = blockIdx.x * 128;

    const v8f z = {0.f, 0.f, 0.f, 0.f, 0.f, 0.f, 0.f, 0.f};
    v8f acc[2][4];
    #pragma unroll
    for (int f = 0; f < 2; ++f)
        #pragma unroll
        for (int g = 0; g < 4; ++g) acc[f][g] = z;

    for (int k0 = 0; k0 < 1024; k0 += 32) {
        __syncthreads();
        if constexpr (MODE == 0) {
            #pragma unroll
            for (int p = 0; p < 4; ++p) {
                int r = p * 32 + (t >> 3);
                int c = (t & 7) * 4;
                float4 fa = *(const float4*)(Af + (size_t)(mBase + r) * 1024 + k0 + c);
                sA[((r * 32 + c) >> 1) + 0] = pk(f2bf(fa.x), f2bf(fa.y));
                sA[((r * 32 + c) >> 1) + 1] = pk(f2bf(fa.z), f2bf(fa.w));
            }
        } else {
            #pragma unroll
            for (int p = 0; p < 2; ++p) {
                int r = p * 64 + (t >> 2);
                int c = (t & 3) * 8;   // bf16 columns, 8 per uint4
                uint4 va = *(const uint4*)(Ah + (size_t)(mBase + r) * 1024 + k0 + c);
                *(uint4*)&sA[(r * 32 + c) >> 1] = va;
            }
        }
        #pragma unroll
        for (int p = 0; p < 4; ++p) {
            int r = p * 32 + (t >> 3);
            int c = (t & 7) * 4;
            float4 fb = *(const float4*)(W + (size_t)(nBase + r) * 1024 + k0 + c);
            sB[((r * 32 + c) >> 1) + 0] = pk(f2bf(fb.x), f2bf(fb.y));
            sB[((r * 32 + c) >> 1) + 1] = pk(f2bf(fb.z), f2bf(fb.w));
        }
        __syncthreads();

        Frag a[2], b[4];
        #pragma unroll
        for (int f = 0; f < 2; ++f) {
            int row = wm * 32 + f * 16 + (lane & 15);
            #pragma unroll
            for (int i = 0; i < 8; ++i)
                a[f].u[i] = sA[(row * 32 + koffs(i, lane)) >> 1];
        }
        #pragma unroll
        for (int g = 0; g < 4; ++g) {
            int row = wn * 64 + g * 16 + (lane & 15);
            #pragma unroll
            for (int i = 0; i < 8; ++i)
                b[g].u[i] = sB[(row * 32 + koffs(i, lane)) >> 1];
        }
        #pragma unroll
        for (int f = 0; f < 2; ++f)
            #pragma unroll
            for (int g = 0; g < 4; ++g)
                acc[f][g] = wmma_bf16(a[f], b[g], acc[f][g]);
    }

    #pragma unroll
    for (int f = 0; f < 2; ++f) {
        #pragma unroll
        for (int g = 0; g < 4; ++g) {
            int col = nBase + wn * 64 + g * 16 + (lane & 15);
            float bval = bias[col];
            #pragma unroll
            for (int j = 0; j < 8; ++j) {
                int row = mBase + wm * 32 + f * 16 + j + ((lane >> 4) << 3);
                float val = acc[f][g][j] + bval;
                if constexpr (MODE == 0) {
                    int bb = row >> 10, nq = row & 1023;
                    int which = col >> 10, cc = col & 1023;
                    int hh = cc >> 6, d = cc & 63;
                    size_t head = (size_t)(bb * 16 + hh);
                    if (which == 0)
                        qo[(head * 1024 + nq) * 64 + d] = f2bf(val * 0.125f);
                    else if (which == 1)
                        ko[(head * 1024 + nq) * 64 + d] = f2bf(val);
                    else
                        vto[(head * 64 + d) * 1024 + nq] = f2bf(val);  // V transposed
                } else {
                    out[(size_t)row * 1024 + col] = val;
                }
            }
        }
    }
}

// ---------------------------------------------------------------------------
// Flash attention: one block = one (b,h) x 128-query tile; 8 waves, each wave
// owns 16 queries. K/V tiles (32 keys) stream through double-buffered LDS via
// TDM (tensor_load_to_lds), pipelined with s_wait_tensorcnt.
// ---------------------------------------------------------------------------
__global__ __launch_bounds__(256) void attn_wmma(
    const unsigned short* __restrict__ qws, const unsigned short* __restrict__ kws,
    const unsigned short* __restrict__ vtws, const float* __restrict__ mask,
    unsigned short* __restrict__ x2)
{
    __shared__ unsigned sQ[128 * 32];          // 128 x 64 bf16
    __shared__ unsigned sK[2][32 * 32];        // 32 keys x 64 dims, double buffered
    __shared__ unsigned sVT[2][64 * 16];       // 64 dims x 32 keys, double buffered
    __shared__ unsigned short sP[8 * 16 * 32]; // per-wave 16x32 P tile
    __shared__ alignas(16) float sM[128 * 32]; // mask tile (fp32)

    const int t = threadIdx.x;
    const int lane = t & 31, wave = t >> 5;
    const int bh = blockIdx.x, qt = blockIdx.y;
    const int b = bh >> 4, h = bh & 15;
    const size_t head = (size_t)bh << 16;      // 1024*64 elements per head
    const int qtb = qt * 128;

    // ---- TDM: stage Q tile + first K/V tiles ----
    if (wave == 0) {
        tdm_load_2d_bf16((unsigned)(uintptr_t)&sQ[0],
                         qws + head + (size_t)qtb * 64, 64, 128, 64);
        tdm_load_2d_bf16((unsigned)(uintptr_t)&sK[0][0], kws + head, 64, 32, 64);
        tdm_load_2d_bf16((unsigned)(uintptr_t)&sVT[0][0], vtws + head, 32, 64, 1024);
        __builtin_amdgcn_s_wait_tensorcnt((short)2);   // Q (oldest) complete
    }
    __syncthreads();

    // resident Q fragments (A-operand layout), K-dim = head dims
    Frag qf[2];
    {
        int row = wave * 16 + (lane & 15);
        #pragma unroll
        for (int kk = 0; kk < 2; ++kk)
            #pragma unroll
            for (int i = 0; i < 8; ++i)
                qf[kk].u[i] = sQ[(row * 64 + kk * 32 + koffs(i, lane)) >> 1];
    }

    const v8f z = {0.f, 0.f, 0.f, 0.f, 0.f, 0.f, 0.f, 0.f};
    v8f o[4];
    #pragma unroll
    for (int n = 0; n < 4; ++n) o[n] = z;
    float mst[8], lst[8];
    #pragma unroll
    for (int j = 0; j < 8; ++j) { mst[j] = -1e30f; lst[j] = 0.f; }

    const int qg0 = qtb + wave * 16 + ((lane >> 4) << 3);  // global q row at vgpr j=0

    for (int it = 0; it < 32; ++it) {
        const int kb = it * 32;
        const int cur = it & 1;
        __syncthreads();   // prior compute done: safe to overwrite next buf + sM

        // ---- stage mask tile (128 x 32 fp32), coalesced float4 ----
        #pragma unroll
        for (int p = 0; p < 4; ++p) {
            int r = p * 32 + (t >> 3);
            int c = (t & 7) * 4;
            *(float4*)&sM[r * 32 + c] =
                *(const float4*)(mask + (size_t)(qtb + r) * 1024 + kb + c);
        }

        // ---- TDM: prefetch next K/V tiles, wait for current ----
        if (wave == 0) {
            if (it + 1 < 32) {
                int nxt = (it + 1) & 1;
                tdm_load_2d_bf16((unsigned)(uintptr_t)&sK[nxt][0],
                                 kws + head + (size_t)(kb + 32) * 64, 64, 32, 64);
                tdm_load_2d_bf16((unsigned)(uintptr_t)&sVT[nxt][0],
                                 vtws + head + (kb + 32), 32, 64, 1024);
                __builtin_amdgcn_s_wait_tensorcnt((short)2);  // tile `it` complete
            } else {
                __builtin_amdgcn_s_wait_tensorcnt((short)0);
            }
        }
        __syncthreads();

        // ---- S = Q @ K^T for this 16q x 32k slice (4 WMMAs) ----
        v8f s[2]; s[0] = z; s[1] = z;
        #pragma unroll
        for (int g = 0; g < 2; ++g) {
            int krow = g * 16 + (lane & 15);
            #pragma unroll
            for (int kk = 0; kk < 2; ++kk) {
                Frag kf;
                #pragma unroll
                for (int i = 0; i < 8; ++i)
                    kf.u[i] = sK[cur][(krow * 64 + kk * 32 + koffs(i, lane)) >> 1];
                s[g] = wmma_bf16(qf[kk], kf, s[g]);
            }
        }

        // ---- add attn_mask from LDS ----
        #pragma unroll
        for (int g = 0; g < 2; ++g) {
            int mc = g * 16 + (lane & 15);
            int mr = wave * 16 + ((lane >> 4) << 3);
            #pragma unroll
            for (int j = 0; j < 8; ++j)
                s[g][j] += sM[(mr + j) * 32 + mc];
        }

        // ---- online softmax (rows live across 16-lane halves) ----
        float al[8];
        #pragma unroll
        for (int j = 0; j < 8; ++j) {
            float cm = fmaxf(s[0][j], s[1][j]);
            cm = fmaxf(cm, __shfl_xor(cm, 1, 32));
            cm = fmaxf(cm, __shfl_xor(cm, 2, 32));
            cm = fmaxf(cm, __shfl_xor(cm, 4, 32));
            cm = fmaxf(cm, __shfl_xor(cm, 8, 32));
            float mn = fmaxf(mst[j], cm);
            float alpha = __expf(mst[j] - mn);
            float p0 = __expf(s[0][j] - mn);
            float p1 = __expf(s[1][j] - mn);
            s[0][j] = p0; s[1][j] = p1;
            float rs = p0 + p1;
            rs += __shfl_xor(rs, 1, 32);
            rs += __shfl_xor(rs, 2, 32);
            rs += __shfl_xor(rs, 4, 32);
            rs += __shfl_xor(rs, 8, 32);
            lst[j] = lst[j] * alpha + rs;
            mst[j] = mn;
            al[j] = alpha;
        }
        #pragma unroll
        for (int n = 0; n < 4; ++n)
            #pragma unroll
            for (int j = 0; j < 8; ++j)
                o[n][j] *= al[j];

        // ---- P: C-layout -> LDS -> A-layout (per-wave region) ----
        const int pBase = wave * 512;
        #pragma unroll
        for (int g = 0; g < 2; ++g)
            #pragma unroll
            for (int j = 0; j < 8; ++j)
                sP[pBase + (j + ((lane >> 4) << 3)) * 32 + g * 16 + (lane & 15)] =
                    f2bf(s[g][j]);
        __syncthreads();

        Frag pf;
        {
            int row = lane & 15;
            #pragma unroll
            for (int i = 0; i < 8; ++i) {
                int idx = pBase + row * 32 + koffs(i, lane);
                pf.u[i] = pk(sP[idx], sP[idx + 1]);
            }
        }

        // ---- O += P @ V (4 WMMAs); V^T tile already in B-operand-friendly form ----
        #pragma unroll
        for (int n = 0; n < 4; ++n) {
            Frag vf;
            int dcol = n * 16 + (lane & 15);
            #pragma unroll
            for (int i = 0; i < 8; ++i)
                vf.u[i] = sVT[cur][(dcol * 32 + koffs(i, lane)) >> 1];
            o[n] = wmma_bf16(pf, vf, o[n]);
        }
    }

    // ---- normalize and store bf16 to (B, N, C) ----
    #pragma unroll
    for (int n = 0; n < 4; ++n) {
        int c = h * 64 + n * 16 + (lane & 15);
        #pragma unroll
        for (int j = 0; j < 8; ++j) {
            float val = o[n][j] * (1.0f / lst[j]);
            x2[((size_t)b * 1024 + (qg0 + j)) * 1024 + c] = f2bf(val);
        }
    }
}

extern "C" void kernel_launch(void* const* d_in, const int* in_sizes, int n_in,
                              void* d_out, int out_size, void* d_ws, size_t ws_size,
                              hipStream_t stream) {
    (void)in_sizes; (void)n_in; (void)out_size; (void)ws_size;
    const float* x    = (const float*)d_in[0];  // (8,1024,1024)
    const float* mask = (const float*)d_in[1];  // (1024,1024)
    const float* win  = (const float*)d_in[2];  // (3072,1024)
    const float* bin  = (const float*)d_in[3];  // (3072,)
    const float* wout = (const float*)d_in[4];  // (1024,1024)
    const float* bout = (const float*)d_in[5];  // (1024,)
    float* out = (float*)d_out;

    const size_t HSZ = (size_t)8 * 16 * 1024 * 64;   // elements per bf16 tensor
    unsigned short* qws  = (unsigned short*)d_ws;    // (B,H,N,D) bf16
    unsigned short* kws  = qws + HSZ;                // (B,H,N,D) bf16
    unsigned short* vtws = kws + HSZ;                // (B,H,D,N) bf16 (transposed)
    unsigned short* x2   = vtws + HSZ;               // (B,N,C)  bf16

    // 1) fused QKV projection (WMMA), scatter bf16 Q/K/V^T
    gemm_wmma<0><<<dim3(3072 / 128, 8192 / 128), 256, 0, stream>>>(
        x, nullptr, win, bin, qws, kws, vtws, nullptr);
    // 2) flash attention (WMMA + TDM double-buffered K/V tiles)
    attn_wmma<<<dim3(128, 8), 256, 0, stream>>>(qws, kws, vtws, mask, x2);
    // 3) output projection (WMMA, bf16 A), fp32 out
    gemm_wmma<1><<<dim3(1024 / 128, 8192 / 128), 256, 0, stream>>>(
        nullptr, x2, wout, bout, nullptr, nullptr, nullptr, out);
}
